// FutureMultiAttentionHead_10325101380398
// MI455X (gfx1250) — matrix operand order repaired and checked by reading, not compiled, not run on hardware
//
#include <hip/hip_runtime.h>

typedef __bf16 bf16_t;
typedef __attribute__((ext_vector_type(16))) __bf16 v16bf;
typedef __attribute__((ext_vector_type(8)))  __bf16 v8bf;
typedef __attribute__((ext_vector_type(8)))  float  v8f;

#define NEG_INF (-__builtin_inff())

// ---------------------------------------------------------------------------
// WMMA helpers (CDNA5 v_wmma_f32_16x16x32_bf16, wave32)
// ---------------------------------------------------------------------------
__device__ inline v8f wmma_bf16(v16bf a, v16bf b, v8f c) {
  return __builtin_amdgcn_wmma_f32_16x16x32_bf16(
      /*neg_a=*/false, a, /*neg_b=*/false, b,
      /*c_mod=*/(short)0, c, /*reuse_a=*/false, /*reuse_b=*/false);
}

__device__ inline v8f vzero8() {
  v8f z;
#pragma unroll
  for (int i = 0; i < 8; ++i) z[i] = 0.0f;
  return z;
}

// A-fragment: 16x32 bf16 tile (row-major, stride lda elements).
// Lane m = lane&15; elems 0-15 <-> K = hlf*16+0..15
// One contiguous 32B run.
__device__ inline v16bf load_a_frag(const bf16_t* a, int lda, int lane) {
  const int m = lane & 15, hlf = lane >> 4;
  v16bf f;
#pragma unroll
  for (int e = 0; e < 16; ++e) {
    f[e]     = a[m * lda + hlf * 16 + e];
  }
  return f;
}

// ---------------------------------------------------------------------------
// Prep kernels: bf16 conversion / transposition into WMMA-friendly layouts
// ---------------------------------------------------------------------------
__global__ __launch_bounds__(256) void cvt_kernel(const float* __restrict__ src,
                                                  bf16_t* __restrict__ dst,
                                                  int ntot) {
  int i = blockIdx.x * 256 + threadIdx.x;
  if (i < ntot) dst[i] = (bf16_t)src[i];
}

// src f32 [1024][N] row-major  ->  dst bf16 [N][1024] (N-major, K-contiguous)
__global__ __launch_bounds__(256) void transpose_w_kernel(
    const float* __restrict__ src, bf16_t* __restrict__ dst, int N) {
  int i = blockIdx.x * 256 + threadIdx.x;   // i = c*1024 + k
  int k = i & 1023, c = i >> 10;
  if (c < N) dst[i] = (bf16_t)src[(size_t)k * N + c];
}

// future_k f32 [16][64][1023] -> fkt bf16 [16][1040][64] (zero pad jf>=1023)
__global__ __launch_bounds__(256) void prep_fkt_kernel(
    const float* __restrict__ fk, bf16_t* __restrict__ fkt) {
  int i = blockIdx.x * 256 + threadIdx.x;   // i = (h*1040 + jf)*64 + d
  int d = i & 63;
  int r = i >> 6;
  int jf = r % 1040;
  int h = r / 1040;
  float v = (jf <= 1022) ? fk[((size_t)h * 64 + d) * 1023 + jf] : 0.0f;
  fkt[i] = (bf16_t)v;
}

// future_v f32 [16][1023][64] -> fvt bf16 [16][64][1040] (zero pad jf>=1023)
__global__ __launch_bounds__(256) void prep_fvt_kernel(
    const float* __restrict__ fv, bf16_t* __restrict__ fvt) {
  int i = blockIdx.x * 256 + threadIdx.x;   // i = (h*64 + hs)*1040 + jf
  int jf = i % 1040;
  int r = i / 1040;
  int hs = r & 63;
  int h = r >> 6;
  float v = (jf <= 1022) ? fv[((size_t)h * 1023 + jf) * 64 + hs] : 0.0f;
  fvt[i] = (bf16_t)v;
}

// ---------------------------------------------------------------------------
// GEMM1: qkv = xb[4096,1024](bf16) @ WqkvT[3072,1024](bf16), LDS-free,
// ping-pong software pipeline (no register rotation copies).
// Scatter: q,k -> [B*H][T][HS] row-major;  v -> [B*H][HS][T] transposed.
// ---------------------------------------------------------------------------
__global__ __launch_bounds__(256) void gemm_qkv_kernel(
    const bf16_t* __restrict__ xb, const bf16_t* __restrict__ wqt,
    bf16_t* __restrict__ qb, bf16_t* __restrict__ kb,
    bf16_t* __restrict__ vbt) {
  const int t = threadIdx.x, lane = t & 31, w = t >> 5;
  const int n = lane & 15, hlf = lane >> 4;
  const int M0 = blockIdx.y * 128, N0 = blockIdx.x * 64;

  v8f acc[4];
#pragma unroll
  for (int nf = 0; nf < 4; ++nf) acc[nf] = vzero8();

  const bf16_t* arow = xb + (size_t)(M0 + w * 16) * 1024;
  const bf16_t* bbase = wqt + (size_t)N0 * 1024 + hlf * 16;

  v16bf aF0 = load_a_frag(arow, 1024, lane);
  v16bf bF0[4], bF1[4];
  v16bf aF1;
#pragma unroll
  for (int nf = 0; nf < 4; ++nf)
    bF0[nf] = *(const v16bf*)(bbase + (size_t)(nf * 16 + n) * 1024);

  for (int kk = 0; kk < 1024; kk += 64) {
    const int k1 = kk + 32;                          // always < 1024
    aF1 = load_a_frag(arow + k1, 1024, lane);
#pragma unroll
    for (int nf = 0; nf < 4; ++nf)
      bF1[nf] = *(const v16bf*)(bbase + (size_t)(nf * 16 + n) * 1024 + k1);
    if (kk + 256 < 1024)
      __builtin_prefetch(arow + (lane & 15) * 1024 + kk + 256, 0, 3);
#pragma unroll
    for (int nf = 0; nf < 4; ++nf) acc[nf] = wmma_bf16(aF0, bF0[nf], acc[nf]);

    const int k2 = (kk + 64 < 1024) ? kk + 64 : 0;   // last: dummy reload
    aF0 = load_a_frag(arow + k2, 1024, lane);
#pragma unroll
    for (int nf = 0; nf < 4; ++nf)
      bF0[nf] = *(const v16bf*)(bbase + (size_t)(nf * 16 + n) * 1024 + k2);
#pragma unroll
    for (int nf = 0; nf < 4; ++nf) acc[nf] = wmma_bf16(aF1, bF1[nf], acc[nf]);
  }

  // ---- epilogue: all selectors are block/wave-uniform ----
  const int which = N0 >> 10;            // 0=q,1=k,2=v
  const int h = (N0 >> 6) & 15;          // head (64-aligned N0)
  const int base_row = M0 + w * 16;
  const int bi = base_row >> 10, ti0 = base_row & 1023;

  if (which == 2) {
    // transposed v: lane's 8 rows are contiguous -> one b128 store per nf
#pragma unroll
    for (int nf = 0; nf < 4; ++nf) {
      v8bf pv;
#pragma unroll
      for (int r = 0; r < 8; ++r) pv[r] = (bf16_t)acc[nf][r];
      bf16_t* p = vbt + (((size_t)bi * 16 + h) * 64 + nf * 16 + n) * 1024 +
                  ti0 + hlf * 8;
      *(v8bf*)p = pv;
    }
  } else {
    bf16_t* dst = (which == 0) ? qb : kb;
    bf16_t* base = dst + (((size_t)bi * 16 + h) * 1024 + ti0 + hlf * 8) * 64;
#pragma unroll
    for (int nf = 0; nf < 4; ++nf)
#pragma unroll
      for (int r = 0; r < 8; ++r)
        base[(size_t)r * 64 + nf * 16 + n] = (bf16_t)acc[nf][r];
  }
}

// ---------------------------------------------------------------------------
// Attention: one wave per 16-row tile per (b,h); flash-style online softmax.
// Causal pass processes 64 score columns per softmax update (4 tiles),
// future band processes its single 32-wide tile.  All WMMA operands are
// direct, aligned 32B vector loads; only P transposes through 2KB of LDS.
// ---------------------------------------------------------------------------
__global__ __launch_bounds__(32) void attn_kernel(
    const bf16_t* __restrict__ qb, const bf16_t* __restrict__ kb,
    const bf16_t* __restrict__ vbt, const bf16_t* __restrict__ fkt,
    const bf16_t* __restrict__ fvt, bf16_t* __restrict__ ao) {
  __shared__ v16bf PlV[64];    // 16x64 bf16 = 2KB
  bf16_t* Pl = (bf16_t*)PlV;

  const int lane = threadIdx.x, n = lane & 15, hlf = lane >> 4;
  const int rt = blockIdx.x, bh = blockIdx.y;
  const int h = bh & 15, b = bh >> 4;
  const float scale = 0.125f;  // 64^-0.5

  const bf16_t* qbase = qb + ((size_t)bh * 1024 + rt * 16) * 64;
  v16bf a_q[2];
  a_q[0] = load_a_frag(qbase, 64, lane);
  a_q[1] = load_a_frag(qbase + 32, 64, lane);

  float mrow[8], lrow[8];
  v8f accO[4];
#pragma unroll
  for (int r = 0; r < 8; ++r) { mrow[r] = NEG_INF; lrow[r] = 0.0f; }
#pragma unroll
  for (int nf = 0; nf < 4; ++nf) accO[nf] = vzero8();

  // ---- causal pass: 4 tiles (64 cols) per online-softmax update ----
  for (int jt4 = 0; jt4 <= rt; jt4 += 4) {
    const int j0 = jt4 * 16;
    // all loads up front: k^T (4 tiles x 2 K-chunks), v^T (4 nf x 2 K-chunks)
    v16bf bk[4][2], bv[4][2];
#pragma unroll
    for (int tt = 0; tt < 4; ++tt) {
      const bf16_t* kr = kb + ((size_t)bh * 1024 + j0 + tt * 16 + n) * 64 +
                         hlf * 16;
      bk[tt][0] = *(const v16bf*)kr;
      bk[tt][1] = *(const v16bf*)(kr + 32);
    }
#pragma unroll
    for (int nf = 0; nf < 4; ++nf) {
      const bf16_t* vr = vbt + ((size_t)bh * 64 + nf * 16 + n) * 1024 + j0 +
                         hlf * 16;
      bv[nf][0] = *(const v16bf*)vr;
      bv[nf][1] = *(const v16bf*)(vr + 32);
    }

    v8f s[4];
#pragma unroll
    for (int tt = 0; tt < 4; ++tt) {
      s[tt] = vzero8();
#pragma unroll
      for (int kc = 0; kc < 2; ++kc) s[tt] = wmma_bf16(a_q[kc], bk[tt][kc], s[tt]);
    }
#pragma unroll
    for (int tt = 0; tt < 4; ++tt)
#pragma unroll
      for (int r = 0; r < 8; ++r) {  // scale + causal mask (j <= i)
        int i = rt * 16 + r + hlf * 8;
        s[tt][r] = (j0 + tt * 16 + n <= i) ? s[tt][r] * scale : NEG_INF;
      }

    // online softmax over 64 columns
    float p[4][8];
#pragma unroll
    for (int r = 0; r < 8; ++r) {
      float vmax = fmaxf(fmaxf(s[0][r], s[1][r]), fmaxf(s[2][r], s[3][r]));
#pragma unroll
      for (int msk = 8; msk >= 1; msk >>= 1)
        vmax = fmaxf(vmax, __shfl_xor(vmax, msk, 32));
      float mnew = fmaxf(mrow[r], vmax);
      float corr = __expf(mrow[r] - mnew);
      mrow[r] = mnew;
      float ls = 0.0f;
#pragma unroll
      for (int tt = 0; tt < 4; ++tt) {
        p[tt][r] = __expf(s[tt][r] - mnew);
        ls += p[tt][r];
      }
#pragma unroll
      for (int msk = 8; msk >= 1; msk >>= 1) ls += __shfl_xor(ls, msk, 32);
      lrow[r] = lrow[r] * corr + ls;
#pragma unroll
      for (int nf = 0; nf < 4; ++nf) accO[nf][r] *= corr;
    }
    // C-layout -> row-major 16x64 LDS for the two P A-fragments
#pragma unroll
    for (int r = 0; r < 8; ++r) {
      int m_loc = r + hlf * 8;
#pragma unroll
      for (int tt = 0; tt < 4; ++tt)
        Pl[m_loc * 64 + tt * 16 + n] = (bf16_t)p[tt][r];
    }
    __syncthreads();
    v16bf ap0 = load_a_frag(Pl, 64, lane);
    v16bf ap1 = load_a_frag(Pl + 32, 64, lane);
#pragma unroll
    for (int nf = 0; nf < 4; ++nf) {
      accO[nf] = wmma_bf16(ap0, bv[nf][0], accO[nf]);
      accO[nf] = wmma_bf16(ap1, bv[nf][1], accO[nf]);
    }
    __syncthreads();
  }

  // ---- future band: one 16x32 tile, column c <-> jf = rt*16 + c ----
  {
    const int jf0 = rt * 16;
    v16bf bk[4], bv[4];
#pragma unroll
    for (int ct = 0; ct < 2; ++ct)
#pragma unroll
      for (int kc = 0; kc < 2; ++kc)
        bk[ct * 2 + kc] = *(const v16bf*)(fkt +
            ((size_t)h * 1040 + jf0 + ct * 16 + n) * 64 + kc * 32 + hlf * 16);
#pragma unroll
    for (int nf = 0; nf < 4; ++nf)
      bv[nf] = *(const v16bf*)(fvt + ((size_t)h * 64 + nf * 16 + n) * 1040 +
                               jf0 + hlf * 16);

    v8f t0 = vzero8(), t1 = vzero8();
#pragma unroll
    for (int kc = 0; kc < 2; ++kc) {
      t0 = wmma_bf16(a_q[kc], bk[kc], t0);
      t1 = wmma_bf16(a_q[kc], bk[2 + kc], t1);
    }
#pragma unroll
    for (int r = 0; r < 8; ++r) {  // band mask: m <= c <= m+15, jf <= 1022
      int m_loc = r + hlf * 8;
      int c0 = n, c1 = 16 + n;
      bool k0 = (c0 >= m_loc) && (jf0 + c0 <= 1022);        // c0<=m+15 auto
      bool k1 = (c1 <= m_loc + 15) && (jf0 + c1 <= 1022);   // c1>=m auto
      t0[r] = k0 ? t0[r] * scale : NEG_INF;
      t1[r] = k1 ? t1[r] * scale : NEG_INF;
    }

    float p0[8], p1[8];
#pragma unroll
    for (int r = 0; r < 8; ++r) {
      float vmax = fmaxf(t0[r], t1[r]);
#pragma unroll
      for (int msk = 8; msk >= 1; msk >>= 1)
        vmax = fmaxf(vmax, __shfl_xor(vmax, msk, 32));
      float mnew = fmaxf(mrow[r], vmax);
      float corr = __expf(mrow[r] - mnew);
      mrow[r] = mnew;
      float e0 = __expf(t0[r] - mnew);
      float e1 = __expf(t1[r] - mnew);
      p0[r] = e0; p1[r] = e1;
      float ls = e0 + e1;
#pragma unroll
      for (int msk = 8; msk >= 1; msk >>= 1) ls += __shfl_xor(ls, msk, 32);
      lrow[r] = lrow[r] * corr + ls;
#pragma unroll
      for (int nf = 0; nf < 4; ++nf) accO[nf][r] *= corr;
    }
#pragma unroll
    for (int r = 0; r < 8; ++r) {
      int m_loc = r + hlf * 8;
      Pl[m_loc * 64 + n]      = (bf16_t)p0[r];
      Pl[m_loc * 64 + 16 + n] = (bf16_t)p1[r];
    }
    __syncthreads();
    v16bf ap = load_a_frag(Pl, 64, lane);
#pragma unroll
    for (int nf = 0; nf < 4; ++nf) accO[nf] = wmma_bf16(ap, bv[nf], accO[nf]);
  }

  // ---- epilogue: O / l -> bf16 attn-out [B,T,C] ----
  bf16_t* abase = ao + ((size_t)b * 1024 + rt * 16 + hlf * 8) * 1024 + h * 64;
#pragma unroll
  for (int nf = 0; nf < 4; ++nf)
#pragma unroll
    for (int r = 0; r < 8; ++r)
      abase[(size_t)r * 1024 + nf * 16 + n] = (bf16_t)(accO[nf][r] / lrow[r]);
}

// ---------------------------------------------------------------------------
// GEMM2: out = ao[4096,1024](bf16) @ WprojT[1024,1024](bf16) -> f32,
// LDS-free, ping-pong pipelined.
// ---------------------------------------------------------------------------
__global__ __launch_bounds__(256) void gemm_proj_kernel(
    const bf16_t* __restrict__ A, const bf16_t* __restrict__ wpt,
    float* __restrict__ Out) {
  const int t = threadIdx.x, lane = t & 31, w = t >> 5;
  const int n = lane & 15, hlf = lane >> 4;
  const int M0 = blockIdx.y * 128, N0 = blockIdx.x * 64;

  v8f acc[4];
#pragma unroll
  for (int nf = 0; nf < 4; ++nf) acc[nf] = vzero8();

  const bf16_t* arow = A + (size_t)(M0 + w * 16) * 1024;
  const bf16_t* bbase = wpt + (size_t)N0 * 1024 + hlf * 16;

  v16bf aF0 = load_a_frag(arow, 1024, lane);
  v16bf bF0[4], bF1[4];
  v16bf aF1;
#pragma unroll
  for (int nf = 0; nf < 4; ++nf)
    bF0[nf] = *(const v16bf*)(bbase + (size_t)(nf * 16 + n) * 1024);

  for (int kk = 0; kk < 1024; kk += 64) {
    const int k1 = kk + 32;
    aF1 = load_a_frag(arow + k1, 1024, lane);
#pragma unroll
    for (int nf = 0; nf < 4; ++nf)
      bF1[nf] = *(const v16bf*)(bbase + (size_t)(nf * 16 + n) * 1024 + k1);
    if (kk + 256 < 1024)
      __builtin_prefetch(arow + (lane & 15) * 1024 + kk + 256, 0, 3);
#pragma unroll
    for (int nf = 0; nf < 4; ++nf) acc[nf] = wmma_bf16(aF0, bF0[nf], acc[nf]);

    const int k2 = (kk + 64 < 1024) ? kk + 64 : 0;
    aF0 = load_a_frag(arow + k2, 1024, lane);
#pragma unroll
    for (int nf = 0; nf < 4; ++nf)
      bF0[nf] = *(const v16bf*)(bbase + (size_t)(nf * 16 + n) * 1024 + k2);
#pragma unroll
    for (int nf = 0; nf < 4; ++nf) acc[nf] = wmma_bf16(aF1, bF1[nf], acc[nf]);
  }

  float* obase = Out + (size_t)(M0 + w * 16 + hlf * 8) * 1024 + N0;
#pragma unroll
  for (int nf = 0; nf < 4; ++nf)
#pragma unroll
    for (int r = 0; r < 8; ++r)
      obase[(size_t)r * 1024 + nf * 16 + n] = acc[nf][r];
}

// ---------------------------------------------------------------------------
extern "C" void kernel_launch(void* const* d_in, const int* in_sizes, int n_in,
                              void* d_out, int out_size, void* d_ws,
                              size_t ws_size, hipStream_t stream) {
  const float* x     = (const float*)d_in[0];  // [4,1024,1024]
  const float* Wqkv  = (const float*)d_in[1];  // [1024,3072]
  const float* Wproj = (const float*)d_in[2];  // [1024,1024]
  const float* fk    = (const float*)d_in[3];  // [16,64,1023]
  const float* fv    = (const float*)d_in[4];  // [16,1023,64]
  float* out = (float*)d_out;                  // [4,1024,1024]

  char* ws = (char*)d_ws;
  size_t off = 0;
  auto take = [&](size_t bytes) {
    void* p = ws + off;
    off += (bytes + 255) & ~(size_t)255;
    return p;
  };
  bf16_t* qb  = (bf16_t*)take((size_t)4 * 1024 * 1024 * 2);  // [B*H][T][HS]
  bf16_t* kb  = (bf16_t*)take((size_t)4 * 1024 * 1024 * 2);  // [B*H][T][HS]
  bf16_t* vbt = (bf16_t*)take((size_t)4 * 1024 * 1024 * 2);  // [B*H][HS][T]
  bf16_t* ao  = (bf16_t*)take((size_t)4 * 1024 * 1024 * 2);  // [B][T][C]
  bf16_t* xb  = (bf16_t*)take((size_t)4 * 1024 * 1024 * 2);  // [4096][1024]
  bf16_t* wqt = (bf16_t*)take((size_t)3 * 1024 * 1024 * 2);  // [3072][1024]
  bf16_t* wpt = (bf16_t*)take((size_t)1024 * 1024 * 2);      // [1024][1024]
  bf16_t* fkt = (bf16_t*)take((size_t)16 * 1040 * 64 * 2);   // [16][1040][64]
  bf16_t* fvt = (bf16_t*)take((size_t)16 * 64 * 1040 * 2);   // [16][64][1040]

  // prep: bf16 conversions + WMMA-friendly transposes
  cvt_kernel<<<16384, 256, 0, stream>>>(x, xb, 4 * 1024 * 1024);
  transpose_w_kernel<<<12288, 256, 0, stream>>>(Wqkv, wqt, 3072);
  transpose_w_kernel<<<4096, 256, 0, stream>>>(Wproj, wpt, 1024);
  prep_fkt_kernel<<<4160, 256, 0, stream>>>(fk, fkt);
  prep_fvt_kernel<<<4160, 256, 0, stream>>>(fv, fvt);

  // main pipeline
  gemm_qkv_kernel<<<dim3(48, 32), 256, 0, stream>>>(xb, wqt, qb, kb, vbt);
  attn_kernel<<<dim3(64, 64), 32, 0, stream>>>(qb, kb, vbt, fkt, fvt, ao);
  gemm_proj_kernel<<<dim3(16, 32), 256, 0, stream>>>(ao, wpt, out);
}
